// FasterRCNN_70265664963203
// MI455X (gfx1250) — matrix operand order repaired and checked by reading, not compiled, not run on hardware
//
#include <hip/hip_runtime.h>
#include <math.h>

typedef __bf16 bf16;
typedef __attribute__((ext_vector_type(8)))  __bf16 v8bf;
typedef __attribute__((ext_vector_type(16))) __bf16 v16bf;
typedef __attribute__((ext_vector_type(8)))  float  v8f;

#define NBATCH   2
#define NA_TOT   112455
#define PX_TOT   37485
#define IMGF     672.0f
#define NEG_BIG  -1000000000.0f
#define BBOX_CLIP_F 4.1351666f   /* log(1000/16) */

// ---------- bf16 helpers (explicit bit handling, RNE) ----------
__device__ inline float bf2f(bf16 b) {
  unsigned short h = __builtin_bit_cast(unsigned short, b);
  unsigned u = ((unsigned)h) << 16;
  return __builtin_bit_cast(float, u);
}
__device__ inline bf16 f2bf(float f) {
  unsigned u = __builtin_bit_cast(unsigned, f);
  u += 0x7fffu + ((u >> 16) & 1u);
  unsigned short h = (unsigned short)(u >> 16);
  return __builtin_bit_cast(bf16, h);
}
__device__ inline bf16 bfzero() { return __builtin_bit_cast(bf16, (unsigned short)0); }
__device__ inline v8bf bz8() {
  v8bf z;
  #pragma unroll
  for (int i = 0; i < 8; ++i) z[i] = bfzero();
  return z;
}
__device__ inline v16bf mk16(v8bf lo, v8bf hi) {
  return __builtin_shufflevector(lo, hi, 0,1,2,3,4,5,6,7,8,9,10,11,12,13,14,15);
}

// ---------------- conversion kernels ----------------
// features NCHW f32 -> NHWC bf16 (per level): dst[(b*PX_TOT + pxoff + px)*256 + c]
__global__ void k_cvt_feat_nhwc(const float* __restrict__ s, bf16* __restrict__ d,
                                int HW, int pxoff) {
  int i = blockIdx.x * blockDim.x + threadIdx.x;   // over B*HW*256
  int tot = NBATCH * HW * 256;
  if (i < tot) {
    int b   = i / (HW * 256);
    int rem = i - b * HW * 256;
    int px  = rem >> 8;
    int c   = rem & 255;
    d[((size_t)b * PX_TOT + pxoff + px) * 256 + c] = f2bf(s[((size_t)b * 256 + c) * HW + px]);
  }
}

// conv weights OIHW f32 -> W^T bf16 [co][tap*256+ci]   (k-ordering: tap major, ci minor)
__global__ void k_cvt_convw(const float* __restrict__ w, bf16* __restrict__ WT) {
  int o = blockIdx.x * blockDim.x + threadIdx.x;   // over 256*2304
  if (o < 256 * 2304) {
    int co  = o / 2304;
    int rem = o - co * 2304;
    int tap = rem >> 8;
    int ci  = rem & 255;
    WT[o] = f2bf(w[(size_t)(co * 256 + ci) * 9 + tap]);
  }
}

// generic [K][N] f32 -> [N][K] bf16 transpose (write-coalesced)
__global__ void k_cvt_transpose(const float* __restrict__ s, bf16* __restrict__ d, int K, int N) {
  int i = blockIdx.x * blockDim.x + threadIdx.x;   // over N*K, dst-linear
  if (i < N * K) {
    int n = i / K;
    int k = i - n * K;
    d[i] = f2bf(s[(size_t)k * N + n]);
  }
}

// ---------------- generic bf16 WMMA GEMM: out = [relu](A[MxK] @ W[KxN] + bias) ----------------
// WT is the transposed weight, [N][K] row-major.
__global__ __launch_bounds__(256) void k_gemm(
    const bf16* __restrict__ A, int lda,
    const bf16* __restrict__ WT,
    const float* __restrict__ bias,
    bf16* __restrict__ outB, float* __restrict__ outF, int ldo,
    int M, int K, int N, int relu)
{
  __shared__ alignas(16) bf16 As[128][40];
  __shared__ alignas(16) bf16 Bs[64][40];
  int tid  = threadIdx.x;
  int lane = tid & 31;
  int w    = tid >> 5;
  int wm   = w & 3, wn = w >> 2;
  int m0 = blockIdx.x * 128;
  int n0 = blockIdx.y * 64;
  int half = lane >> 4;
  int l15  = lane & 15;

  v8f z = {0.f,0.f,0.f,0.f,0.f,0.f,0.f,0.f};
  v8f acc[2][2];
  acc[0][0] = z; acc[0][1] = z; acc[1][0] = z; acc[1][1] = z;

  int arow = tid & 127;
  int acb  = (tid >> 7) * 16;
  int brow = tid & 63;
  int bkb  = (tid >> 6) * 8;

  for (int k0 = 0; k0 < K; k0 += 32) {
    { // A tile 128x32: two 16B vector loads per thread (uniform row guard)
      int gm = m0 + arow;
      v8bf v0 = bz8(), v1 = bz8();
      if (gm < M) {
        const v8bf* src = (const v8bf*)(A + (size_t)gm * lda + (k0 + acb));
        v0 = src[0]; v1 = src[1];
      }
      *(v8bf*)&As[arow][acb]     = v0;
      *(v8bf*)&As[arow][acb + 8] = v1;
    }
    { // B tile: WT rows are contiguous in K -> one 16B vector load per thread
      int gn = n0 + brow;
      v8bf v0 = bz8();
      if (gn < N) v0 = *(const v8bf*)(WT + (size_t)gn * K + (k0 + bkb));
      *(v8bf*)&Bs[brow][bkb] = v0;
    }
    __syncthreads();

    v16bf a[2], bfr[2];
    {
      int r0 = wm * 32 + l15;
      a[0] = mk16(*(const v8bf*)&As[r0][half * 8],      *(const v8bf*)&As[r0][16 + half * 8]);
      a[1] = mk16(*(const v8bf*)&As[r0 + 16][half * 8], *(const v8bf*)&As[r0 + 16][16 + half * 8]);
      int c0 = wn * 32 + l15;
      bfr[0] = mk16(*(const v8bf*)&Bs[c0][half * 8],      *(const v8bf*)&Bs[c0][16 + half * 8]);
      bfr[1] = mk16(*(const v8bf*)&Bs[c0 + 16][half * 8], *(const v8bf*)&Bs[c0 + 16][16 + half * 8]);
    }
    #pragma unroll
    for (int ti = 0; ti < 2; ++ti)
      #pragma unroll
      for (int tj = 0; tj < 2; ++tj)
        acc[ti][tj] = __builtin_amdgcn_wmma_f32_16x16x32_bf16(
            false, a[ti], false, bfr[tj], (short)0, acc[ti][tj], false, false);
    __syncthreads();
  }

  #pragma unroll
  for (int ti = 0; ti < 2; ++ti)
    #pragma unroll
    for (int tj = 0; tj < 2; ++tj) {
      int gn = n0 + wn * 32 + tj * 16 + l15;
      #pragma unroll
      for (int r = 0; r < 8; ++r) {
        int gm = m0 + wm * 32 + ti * 16 + half * 8 + r;
        if (gm < M && gn < N) {
          float v = acc[ti][tj][r] + bias[gn];
          if (relu) v = fmaxf(v, 0.f);
          if (outF) outF[(size_t)gm * ldo + gn] = v;
          else      outB[(size_t)gm * ldo + gn] = f2bf(v);
        }
      }
    }
}

// ---------------- 3x3 conv as implicit GEMM, NHWC features ----------------
// K ordered tap-major: k = tap*256 + ci; one K-step (32) stays inside one tap,
// so per-row bounds are uniform and the channel slice is contiguous (2x16B loads).
__global__ __launch_bounds__(256) void k_conv3(
    const bf16* __restrict__ fbf, int pxoff,   // NHWC: [(b*PX_TOT + pxoff + px)*256 + c]
    const bf16* __restrict__ WT,               // [256][2304]
    const float* __restrict__ bias,            // [256]
    bf16* __restrict__ t,
    int H, int W)
{
  __shared__ alignas(16) bf16 As[128][40];
  __shared__ alignas(16) bf16 Bs[64][40];
  int b = blockIdx.z;
  const bf16* fin = fbf + ((size_t)b * PX_TOT + pxoff) * 256;
  bf16* out = t + ((size_t)b * PX_TOT + pxoff) * 256;
  int HW = H * W;

  int tid  = threadIdx.x;
  int lane = tid & 31;
  int w    = tid >> 5;
  int wm   = w & 3, wn = w >> 2;
  int m0 = blockIdx.x * 128;
  int n0 = blockIdx.y * 64;
  int half = lane >> 4;
  int l15  = lane & 15;

  v8f z = {0.f,0.f,0.f,0.f,0.f,0.f,0.f,0.f};
  v8f acc[2][2];
  acc[0][0] = z; acc[0][1] = z; acc[1][0] = z; acc[1][1] = z;

  int arow = tid & 127;
  int acb  = (tid >> 7) * 16;
  int brow = tid & 63;
  int bkb  = (tid >> 6) * 8;

  int m = m0 + arow;
  int y = m / W, x = m - y * W;

  for (int k0 = 0; k0 < 2304; k0 += 32) {
    int tap = k0 >> 8;
    int c0  = k0 & 255;
    int dy  = tap / 3 - 1;
    int dx  = tap - (tap / 3) * 3 - 1;
    { // A tile: NHWC gather, uniform guard per (row, tap)
      int yy = y + dy, xx = x + dx;
      v8bf v0 = bz8(), v1 = bz8();
      if (m < HW && yy >= 0 && yy < H && xx >= 0 && xx < W) {
        const v8bf* src = (const v8bf*)(fin + ((size_t)(yy * W + xx)) * 256 + (c0 + acb));
        v0 = src[0]; v1 = src[1];
      }
      *(v8bf*)&As[arow][acb]     = v0;
      *(v8bf*)&As[arow][acb + 8] = v1;
    }
    { // weights: contiguous in k
      int gn = n0 + brow;
      *(v8bf*)&Bs[brow][bkb] = *(const v8bf*)(WT + (size_t)gn * 2304 + (k0 + bkb));
    }
    __syncthreads();

    v16bf a[2], bfr[2];
    {
      int r0 = wm * 32 + l15;
      a[0] = mk16(*(const v8bf*)&As[r0][half * 8],      *(const v8bf*)&As[r0][16 + half * 8]);
      a[1] = mk16(*(const v8bf*)&As[r0 + 16][half * 8], *(const v8bf*)&As[r0 + 16][16 + half * 8]);
      int c1 = wn * 32 + l15;
      bfr[0] = mk16(*(const v8bf*)&Bs[c1][half * 8],      *(const v8bf*)&Bs[c1][16 + half * 8]);
      bfr[1] = mk16(*(const v8bf*)&Bs[c1 + 16][half * 8], *(const v8bf*)&Bs[c1 + 16][16 + half * 8]);
    }
    #pragma unroll
    for (int ti = 0; ti < 2; ++ti)
      #pragma unroll
      for (int tj = 0; tj < 2; ++tj)
        acc[ti][tj] = __builtin_amdgcn_wmma_f32_16x16x32_bf16(
            false, a[ti], false, bfr[tj], (short)0, acc[ti][tj], false, false);
    __syncthreads();
  }

  #pragma unroll
  for (int ti = 0; ti < 2; ++ti)
    #pragma unroll
    for (int tj = 0; tj < 2; ++tj) {
      int gn = n0 + wn * 32 + tj * 16 + l15;
      #pragma unroll
      for (int r = 0; r < 8; ++r) {
        int gm = m0 + wm * 32 + ti * 16 + half * 8 + r;
        if (gm < HW) {
          float v = acc[ti][tj][r] + bias[gn];
          v = fmaxf(v, 0.f);
          out[(size_t)gm * 256 + gn] = f2bf(v);
        }
      }
    }
}

// ---------------- fused 1x1 RPN heads (3 logits + 12 deltas per pixel) ----------------
__global__ __launch_bounds__(256) void k_rpn_heads(
    const bf16* __restrict__ t, int pxoff, int aoff, int HW,
    const float* __restrict__ cls_w, const float* __restrict__ cls_b,
    const float* __restrict__ bbox_w, const float* __restrict__ bbox_b,
    float* __restrict__ obj, float* __restrict__ dl)
{
  __shared__ float wc[3 * 256];
  __shared__ float wb[12 * 256];
  int tid = threadIdx.x;
  for (int i = tid; i < 3 * 256; i += 256)  wc[i] = cls_w[i];
  for (int i = tid; i < 12 * 256; i += 256) wb[i] = bbox_w[i];
  __syncthreads();
  int b  = blockIdx.y;
  int px = blockIdx.x * 256 + tid;
  if (px >= HW) return;
  const v8bf* tp = (const v8bf*)(t + ((size_t)b * PX_TOT + pxoff + px) * 256);
  float accC[3]  = {0.f, 0.f, 0.f};
  float accB[12] = {0.f,0.f,0.f,0.f,0.f,0.f,0.f,0.f,0.f,0.f,0.f,0.f};
  for (int c8 = 0; c8 < 32; ++c8) {
    v8bf vv = tp[c8];
    #pragma unroll
    for (int i = 0; i < 8; ++i) {
      int c = c8 * 8 + i;
      float tv = bf2f(vv[i]);
      #pragma unroll
      for (int a2 = 0; a2 < 3; ++a2)  accC[a2] += tv * wc[a2 * 256 + c];
      #pragma unroll
      for (int j = 0; j < 12; ++j)    accB[j]  += tv * wb[j * 256 + c];
    }
  }
  size_t base = (size_t)b * NA_TOT + aoff + (size_t)px * 3;
  #pragma unroll
  for (int a2 = 0; a2 < 3; ++a2) {
    obj[base + a2] = accC[a2] + cls_b[a2];
    #pragma unroll
    for (int j = 0; j < 4; ++j)
      dl[(base + a2) * 4 + j] = accB[a2 * 4 + j] + bbox_b[a2 * 4 + j];
  }
}

// ---------------- anchor decode + clip; also stage destructible score copy ----------------
__global__ void k_decode(const float* __restrict__ obj, const float* __restrict__ dl,
                         float* __restrict__ boxes, float* __restrict__ sc)
{
  int idx = blockIdx.x * blockDim.x + threadIdx.x;
  if (idx >= NBATCH * NA_TOT) return;
  int i = idx % NA_TOT;
  int off, W, stride; float size;
  if (i < 84672)       { off = 0;      W = 168; stride = 4;  size = 32.f;  }
  else if (i < 105840) { off = 84672;  W = 84;  stride = 8;  size = 64.f;  }
  else if (i < 111132) { off = 105840; W = 42;  stride = 16; size = 128.f; }
  else                 { off = 111132; W = 21;  stride = 32; size = 256.f; }
  int loc = i - off;
  int a = loc % 3;
  int px = loc / 3;
  int y = px / W, x = px - y * W;
  float hr = (a == 0) ? 0.7071067811865476f : (a == 1 ? 1.0f : 1.4142135623730951f);
  float ws = size / hr, hs = size * hr;
  float bx0 = rintf(-0.5f * ws), by0 = rintf(-0.5f * hs);
  float bx1 = rintf( 0.5f * ws), by1 = rintf( 0.5f * hs);
  float sx = (float)(x * stride), sy = (float)(y * stride);
  float ax0 = sx + bx0, ay0 = sy + by0, ax1 = sx + bx1, ay1 = sy + by1;
  float wa = ax1 - ax0, ha = ay1 - ay0;
  float cx = ax0 + 0.5f * wa, cy = ay0 + 0.5f * ha;
  const float* d = dl + (size_t)idx * 4;
  float ddx = d[0], ddy = d[1];
  float ddw = fminf(d[2], BBOX_CLIP_F), ddh = fminf(d[3], BBOX_CLIP_F);
  float pcx = ddx * wa + cx, pcy = ddy * ha + cy;
  float pw = expf(ddw) * wa, ph = expf(ddh) * ha;
  float* bo = boxes + (size_t)idx * 4;
  bo[0] = fminf(fmaxf(pcx - 0.5f * pw, 0.f), IMGF);
  bo[1] = fminf(fmaxf(pcy - 0.5f * ph, 0.f), IMGF);
  bo[2] = fminf(fmaxf(pcx + 0.5f * pw, 0.f), IMGF);
  bo[3] = fminf(fmaxf(pcy + 0.5f * ph, 0.f), IMGF);
  sc[idx] = obj[idx];
}

// ---------------- per-level top-1000 (iterative argmax, stable ties) ----------------
__global__ __launch_bounds__(1024) void k_topk(float* __restrict__ sc,
                                               float* __restrict__ pickV, int* __restrict__ pickI)
{
  int l = blockIdx.x, b = blockIdx.y;
  int n, off;
  if (l == 0)      { n = 84672; off = 0;      }
  else if (l == 1) { n = 21168; off = 84672;  }
  else if (l == 2) { n = 5292;  off = 105840; }
  else             { n = 1323;  off = 111132; }
  float* s = sc + (size_t)b * NA_TOT + off;
  __shared__ float rv[1024];
  __shared__ int   ri[1024];
  int tid = threadIdx.x;
  for (int it = 0; it < 1000; ++it) {
    float bv = -INFINITY; int bi = 0x7fffffff;
    for (int j = tid; j < n; j += 1024) {
      float v = s[j];
      if (v > bv) { bv = v; bi = j; }
    }
    rv[tid] = bv; ri[tid] = bi;
    __syncthreads();
    for (int st = 512; st > 0; st >>= 1) {
      if (tid < st) {
        float ov = rv[tid + st]; int oi = ri[tid + st];
        if (ov > rv[tid] || (ov == rv[tid] && oi < ri[tid])) { rv[tid] = ov; ri[tid] = oi; }
      }
      __syncthreads();
    }
    if (tid == 0) {
      pickV[(size_t)b * 4000 + l * 1000 + it] = rv[0];
      pickI[(size_t)b * 4000 + l * 1000 + it] = off + ri[0];
      s[ri[0]] = -INFINITY;
    }
    __syncthreads();
  }
}

// ---------------- concat + filter + stable descending sort (bitonic on (score,pos)) ----------------
__global__ __launch_bounds__(512) void k_sort(
    const float* __restrict__ pickV, const int* __restrict__ pickI,
    const float* __restrict__ boxes,
    float* __restrict__ sScore, float* __restrict__ sBox, float* __restrict__ sObox)
{
  int b = blockIdx.x;
  __shared__ float sk[4096];
  __shared__ int   sp[4096];
  int tid = threadIdx.x;
  for (int i = tid; i < 4096; i += 512) {
    float key = -3.4e38f;
    if (i < 4000) {
      int gi = pickI[b * 4000 + i];
      const float* bo = boxes + ((size_t)b * NA_TOT + gi) * 4;
      float wv = bo[2] - bo[0], hv = bo[3] - bo[1];
      float s = pickV[b * 4000 + i];
      if (!(wv >= 0.001f && hv >= 0.001f)) s = NEG_BIG;
      key = s;
    }
    sk[i] = key; sp[i] = i;
  }
  for (int k = 2; k <= 4096; k <<= 1)
    for (int j = k >> 1; j > 0; j >>= 1) {
      __syncthreads();
      for (int i = tid; i < 4096; i += 512) {
        int p = i ^ j;
        if (p > i) {
          bool up = ((i & k) == 0);
          float ka = sk[i], kb = sk[p];
          int pa = sp[i], pb = sp[p];
          bool aFirst = (ka > kb) || (ka == kb && pa < pb);
          if (up ? !aFirst : aFirst) { sk[i] = kb; sk[p] = ka; sp[i] = pb; sp[p] = pa; }
        }
      }
    }
  __syncthreads();
  for (int i = tid; i < 4000; i += 512) {
    int pos = sp[i];
    int gi = pickI[b * 4000 + pos];
    const float* bo = boxes + ((size_t)b * NA_TOT + gi) * 4;
    float lvl = (float)(pos / 1000);
    sScore[b * 4000 + i] = sk[i];
    #pragma unroll
    for (int j2 = 0; j2 < 4; ++j2) {
      float v = bo[j2];
      sBox [((size_t)b * 4000 + i) * 4 + j2] = v;
      sObox[((size_t)b * 4000 + i) * 4 + j2] = v + lvl * (2.0f * IMGF);
    }
  }
}

// ---------------- greedy NMS over sorted, level-offset boxes ----------------
__global__ __launch_bounds__(1024) void k_nms(const float* __restrict__ sObox, int* __restrict__ supp)
{
  int b = blockIdx.x;
  const float* B_ = sObox + (size_t)b * 4000 * 4;
  int* sp = supp + b * 4000;
  int tid = threadIdx.x;
  for (int i = tid; i < 4000; i += 1024) sp[i] = 0;
  __syncthreads();
  for (int i = 0; i < 3999; ++i) {
    if (sp[i] == 0) {
      float ax0 = B_[i*4], ay0 = B_[i*4+1], ax1 = B_[i*4+2], ay1 = B_[i*4+3];
      float areaA = (ax1 - ax0) * (ay1 - ay0);
      for (int j = i + 1 + tid; j < 4000; j += 1024) {
        float bx0 = B_[j*4], by0 = B_[j*4+1], bx1 = B_[j*4+2], by1 = B_[j*4+3];
        float areaB = (bx1 - bx0) * (by1 - by0);
        float ix = fminf(ax1, bx1) - fmaxf(ax0, bx0);
        float iy = fminf(ay1, by1) - fmaxf(ay0, by0);
        float inter = fmaxf(ix, 0.f) * fmaxf(iy, 0.f);
        float iou = inter / (areaA + areaB - inter + 1e-9f);
        if (iou > 0.7f) sp[j] = 1;
      }
    }
    __syncthreads();
  }
}

// ---------------- keep selection (top_k(where(supp,-1e9,scr)) semantics) + proposal gather ----------------
__global__ __launch_bounds__(256) void k_keep(
    const float* __restrict__ sScore, const float* __restrict__ sBox,
    const int* __restrict__ supp,
    float* __restrict__ prop, float* __restrict__ outProp)
{
  int b = blockIdx.x;
  __shared__ int keep[1000];
  if (threadIdx.x == 0) {
    int cnt = 0;
    for (int i = 0; i < 4000 && cnt < 1000; ++i) {
      float v = supp[b * 4000 + i] ? NEG_BIG : sScore[b * 4000 + i];
      if (v > NEG_BIG) keep[cnt++] = i;
    }
    for (int i = 0; i < 4000 && cnt < 1000; ++i) {
      float v = supp[b * 4000 + i] ? NEG_BIG : sScore[b * 4000 + i];
      if (v <= NEG_BIG) keep[cnt++] = i;
    }
  }
  __syncthreads();
  for (int r = threadIdx.x; r < 1000; r += 256) {
    int i = keep[r];
    #pragma unroll
    for (int j = 0; j < 4; ++j) {
      float v = sBox[((size_t)b * 4000 + i) * 4 + j];
      prop   [((size_t)b * 1000 + r) * 4 + j] = v;
      outProp[((size_t)b * 1000 + r) * 4 + j] = v;
    }
  }
}

// ---------------- ROI align (7x7, SR=2) -> bf16 box-feature rows ----------------
__global__ __launch_bounds__(256) void k_roialign(
    const float* __restrict__ f0, const float* __restrict__ f1,
    const float* __restrict__ f2, const float* __restrict__ f3,
    const float* __restrict__ prop, bf16* __restrict__ xa)
{
  int roi = blockIdx.x, b = blockIdx.y, c = threadIdx.x;
  const float* bo = prop + ((size_t)b * 1000 + roi) * 4;
  float x0 = bo[0], y0 = bo[1], x1 = bo[2], y1 = bo[3];
  float area = fmaxf(x1 - x0, 0.f) * fmaxf(y1 - y0, 0.f);
  float kf = floorf(4.f + log2f(sqrtf(area) / 224.f + 1e-6f));
  kf = fminf(fmaxf(kf, 2.f), 5.f);
  int lvl = (int)kf - 2;
  const float* f; int H, stride;
  if (lvl == 0)      { f = f0; H = 168; stride = 4;  }
  else if (lvl == 1) { f = f1; H = 84;  stride = 8;  }
  else if (lvl == 2) { f = f2; H = 42;  stride = 16; }
  else               { f = f3; H = 21;  stride = 32; }
  int W = H;
  float scale = 1.f / (float)stride;
  const float* plane = f + (size_t)(b * 256 + c) * H * W;
  float rx0 = x0 * scale, ry0 = y0 * scale, rx1 = x1 * scale, ry1 = y1 * scale;
  float rw = fmaxf(rx1 - rx0, 1.f), rh = fmaxf(ry1 - ry0, 1.f);
  float sxw = rw / 7.f, syh = rh / 7.f;
  bf16* orow = xa + (size_t)(b * 1000 + roi) * 12544 + (size_t)c * 49;
  for (int py = 0; py < 7; ++py) {
    for (int px = 0; px < 7; ++px) {
      float sum = 0.f;
      #pragma unroll
      for (int sy = 0; sy < 2; ++sy) {
        float yv = ry0 + ((float)(py * 2 + sy) + 0.5f) * 0.5f * syh;
        yv = fminf(fmaxf(yv, 0.f), (float)(H - 1));
        int yi0 = (int)floorf(yv);
        int yi1 = yi0 + 1; if (yi1 > H - 1) yi1 = H - 1;
        float ly = yv - (float)yi0;
        #pragma unroll
        for (int sx = 0; sx < 2; ++sx) {
          float xv = rx0 + ((float)(px * 2 + sx) + 0.5f) * 0.5f * sxw;
          xv = fminf(fmaxf(xv, 0.f), (float)(W - 1));
          int xi0 = (int)floorf(xv);
          int xi1 = xi0 + 1; if (xi1 > W - 1) xi1 = W - 1;
          float lx = xv - (float)xi0;
          float v00 = plane[yi0 * W + xi0], v01 = plane[yi0 * W + xi1];
          float v10 = plane[yi1 * W + xi0], v11 = plane[yi1 * W + xi1];
          sum += v00 * (1.f - ly) * (1.f - lx) + v01 * (1.f - ly) * lx
               + v10 * ly * (1.f - lx)         + v11 * ly * lx;
        }
      }
      orow[py * 7 + px] = f2bf(sum * 0.25f);
    }
  }
}

// =======================================================================
extern "C" void kernel_launch(void* const* d_in, const int* in_sizes, int n_in,
                              void* d_out, int out_size, void* d_ws, size_t ws_size,
                              hipStream_t stream) {
  (void)in_sizes; (void)n_in; (void)out_size; (void)ws_size;
  const float* f[4]    = {(const float*)d_in[0], (const float*)d_in[1],
                          (const float*)d_in[2], (const float*)d_in[3]};
  const float* conv_w  = (const float*)d_in[4];
  const float* conv_b  = (const float*)d_in[5];
  const float* cls_w   = (const float*)d_in[6];
  const float* cls_b   = (const float*)d_in[7];
  const float* bbox_w  = (const float*)d_in[8];
  const float* bbox_b  = (const float*)d_in[9];
  const float* fc6_w   = (const float*)d_in[10];
  const float* fc6_b   = (const float*)d_in[11];
  const float* fc7_w   = (const float*)d_in[12];
  const float* fc7_b   = (const float*)d_in[13];
  const float* pcls_w  = (const float*)d_in[14];
  const float* pcls_b  = (const float*)d_in[15];
  const float* pbb_w   = (const float*)d_in[16];
  const float* pbb_b   = (const float*)d_in[17];

  char* wsB = (char*)d_ws;
  size_t off = 0;
  auto alloc = [&](size_t bytes) -> void* {
    void* p = wsB + off;
    off += (bytes + 255) & ~(size_t)255;
    return p;
  };
  bf16*  fbf    = (bf16*) alloc((size_t)NBATCH * PX_TOT * 256 * 2);   // NHWC bf16 features
  bf16*  convWT = (bf16*) alloc((size_t)256 * 2304 * 2);              // [co][k]
  bf16*  fc6T   = (bf16*) alloc((size_t)1024 * 12544 * 2);            // [n][k]
  bf16*  fc7T   = (bf16*) alloc((size_t)1024 * 1024 * 2);
  bf16*  pclsT  = (bf16*) alloc((size_t)91 * 1024 * 2);
  bf16*  pbbT   = (bf16*) alloc((size_t)364 * 1024 * 2);
  bf16*  t      = (bf16*) alloc((size_t)NBATCH * PX_TOT * 256 * 2);
  float* obj    = (float*)alloc((size_t)NBATCH * NA_TOT * 4);
  float* dl     = (float*)alloc((size_t)NBATCH * NA_TOT * 4 * 4);
  float* boxes  = (float*)alloc((size_t)NBATCH * NA_TOT * 4 * 4);
  float* scS    = (float*)alloc((size_t)NBATCH * NA_TOT * 4);
  float* pickV  = (float*)alloc((size_t)NBATCH * 4000 * 4);
  int*   pickI  = (int*)  alloc((size_t)NBATCH * 4000 * 4);
  float* sScore = (float*)alloc((size_t)NBATCH * 4000 * 4);
  float* sBox   = (float*)alloc((size_t)NBATCH * 4000 * 4 * 4);
  float* sObox  = (float*)alloc((size_t)NBATCH * 4000 * 4 * 4);
  int*   supp   = (int*)  alloc((size_t)NBATCH * 4000 * 4);
  float* prop   = (float*)alloc((size_t)NBATCH * 1000 * 4 * 4);
  bf16*  xa     = (bf16*) alloc((size_t)2000 * 12544 * 2);
  bf16*  x1     = (bf16*) alloc((size_t)2000 * 1024 * 2);
  bf16*  x2     = (bf16*) alloc((size_t)2000 * 1024 * 2);

  const int HWl[4]    = {168, 84, 42, 21};
  const int HW2l[4]   = {28224, 7056, 1764, 441};
  const int pxoffl[4] = {0, 28224, 35280, 37044};
  const int aoffl[4]  = {0, 84672, 105840, 111132};

  // 1) precision conversion + weight transposes
  for (int l = 0; l < 4; ++l) {
    int n = NBATCH * HW2l[l] * 256;
    k_cvt_feat_nhwc<<<dim3((n + 255) / 256), dim3(256), 0, stream>>>(f[l], fbf, HW2l[l], pxoffl[l]);
  }
  k_cvt_convw<<<dim3((256 * 2304 + 255) / 256), dim3(256), 0, stream>>>(conv_w, convWT);
  k_cvt_transpose<<<dim3((12544 * 1024 + 255) / 256), dim3(256), 0, stream>>>(fc6_w, fc6T, 12544, 1024);
  k_cvt_transpose<<<dim3((1024 * 1024 + 255) / 256),  dim3(256), 0, stream>>>(fc7_w, fc7T, 1024, 1024);
  k_cvt_transpose<<<dim3((1024 * 91 + 255) / 256),    dim3(256), 0, stream>>>(pcls_w, pclsT, 1024, 91);
  k_cvt_transpose<<<dim3((1024 * 364 + 255) / 256),   dim3(256), 0, stream>>>(pbb_w, pbbT, 1024, 364);

  // 2) RPN 3x3 conv (implicit WMMA GEMM) + heads
  for (int l = 0; l < 4; ++l) {
    dim3 g((HW2l[l] + 127) / 128, 4, NBATCH);
    k_conv3<<<g, dim3(256), 0, stream>>>(fbf, pxoffl[l], convWT, conv_b, t, HWl[l], HWl[l]);
  }
  for (int l = 0; l < 4; ++l) {
    dim3 g((HW2l[l] + 255) / 256, NBATCH);
    k_rpn_heads<<<g, dim3(256), 0, stream>>>(t, pxoffl[l], aoffl[l], HW2l[l],
                                             cls_w, cls_b, bbox_w, bbox_b, obj, dl);
  }

  // 3) proposals
  k_decode<<<dim3((NBATCH * NA_TOT + 255) / 256), dim3(256), 0, stream>>>(obj, dl, boxes, scS);
  k_topk<<<dim3(4, NBATCH), dim3(1024), 0, stream>>>(scS, pickV, pickI);
  k_sort<<<dim3(NBATCH), dim3(512), 0, stream>>>(pickV, pickI, boxes, sScore, sBox, sObox);
  k_nms<<<dim3(NBATCH), dim3(1024), 0, stream>>>(sObox, supp);
  float* outProp = (float*)d_out + 2000 * 91 + 2000 * 364;
  k_keep<<<dim3(NBATCH), dim3(256), 0, stream>>>(sScore, sBox, supp, prop, outProp);

  // 4) ROI align -> box features (bf16)
  k_roialign<<<dim3(1000, NBATCH), dim3(256), 0, stream>>>(f[0], f[1], f[2], f[3], prop, xa);

  // 5) box head (WMMA GEMMs)
  k_gemm<<<dim3(16, 16), dim3(256), 0, stream>>>(xa, 12544, fc6T, fc6_b,
                                                 x1, (float*)nullptr, 1024, 2000, 12544, 1024, 1);
  k_gemm<<<dim3(16, 16), dim3(256), 0, stream>>>(x1, 1024, fc7T, fc7_b,
                                                 x2, (float*)nullptr, 1024, 2000, 1024, 1024, 1);
  k_gemm<<<dim3(16, 2), dim3(256), 0, stream>>>(x2, 1024, pclsT, pcls_b,
                                                (bf16*)nullptr, (float*)d_out, 91, 2000, 1024, 91, 0);
  k_gemm<<<dim3(16, 6), dim3(256), 0, stream>>>(x2, 1024, pbbT, pbb_b,
                                                (bf16*)nullptr, (float*)d_out + 2000 * 91, 364, 2000, 1024, 364, 0);
}
